// DiseaseClassifier_29807073034376
// MI455X (gfx1250) — compile-verified, matrix operand order checked by reading, not run on hardware
//
#include <hip/hip_runtime.h>
#include <math.h>

#define Bdim 4096
#define Rdim 29
#define Hdim 768
#define Ddim 14
#define Kdim 384
#define LN_EPS 1e-5f

#define AP  (Hdim + 8)   // LDS A-tile row pitch (ushorts): stride 388 dwords -> conflict-free b128 reads
#define HP  (Kdim + 8)   // LDS h-buffer row pitch (floats)

typedef __attribute__((ext_vector_type(16))) __bf16 v16bf;
typedef __attribute__((ext_vector_type(8)))  float v8f;

// round-to-nearest-even float -> bf16 bit pattern
__device__ __forceinline__ unsigned short f2bf(float f) {
    unsigned int u = __float_as_uint(f);
    unsigned int r = u + 0x7fffu + ((u >> 16) & 1u);
    return (unsigned short)(r >> 16);
}

// ---------------- Kernel 0a: per-disease region counts ----------------
__global__ void counts_kernel(const int* __restrict__ mask,
                              float* __restrict__ counts,
                              float* __restrict__ rdenom) {
    int d = threadIdx.x;
    if (d < Ddim) {
        int c = 0;
        for (int r = 0; r < Rdim; ++r) c += mask[r * Ddim + d];
        counts[d] = (float)c;
        rdenom[d] = 1.0f / fmaxf((float)c, 1.0f);
    }
}

// ---------------- Kernel 0b: W1 [D,H,K] f32 -> W1T [D,K,H] bf16 -------
__global__ __launch_bounds__(256)
void w1t_kernel(const float* __restrict__ W1, unsigned short* __restrict__ w1t) {
    const int dk = blockIdx.x;              // 0 .. D*K-1
    const int d = dk / Kdim, k = dk % Kdim;
    for (int h = threadIdx.x; h < Hdim; h += 256)
        w1t[(size_t)dk * Hdim + h] = f2bf(W1[((size_t)d * Hdim + h) * Kdim + k]);
}

// ---------------- Kernel 1: masked mean pool, write pooledT [D,B,H] bf16
__global__ __launch_bounds__(256)
void pool_kernel(const float* __restrict__ rf, const int* __restrict__ mask,
                 const float* __restrict__ rdenom,
                 unsigned short* __restrict__ pooledT) {
    const int b = blockIdx.y;
    const int h = blockIdx.x * 256 + threadIdx.x;   // H = 3*256
    __shared__ float sm[Rdim * Ddim];
    __shared__ float srd[Ddim];
    for (int i = threadIdx.x; i < Rdim * Ddim; i += 256) sm[i] = (float)mask[i];
    if (threadIdx.x < Ddim) srd[threadIdx.x] = rdenom[threadIdx.x];
    __syncthreads();

    float sums[Ddim];
#pragma unroll
    for (int d = 0; d < Ddim; ++d) sums[d] = 0.0f;

    const float* src = rf + (size_t)b * Rdim * Hdim + h;
    for (int r = 0; r < Rdim; ++r) {
        float val = src[(size_t)r * Hdim];          // coalesced across threads
#pragma unroll
        for (int d = 0; d < Ddim; ++d) sums[d] += sm[r * Ddim + d] * val;
    }
#pragma unroll
    for (int d = 0; d < Ddim; ++d)
        pooledT[((size_t)d * Bdim + b) * Hdim + h] = f2bf(sums[d] * srd[d]);
}

// ---------------- Kernel 2: bf16 WMMA GEMM + fused LN/GELU/dot --------
// Block = 64 batch rows x full K=384. 8 waves; wave w owns cols 48w..48w+47.
// A tile staged once in LDS (shared by all waves); B streamed from L2.
union SmemU {
    unsigned short a[64 * AP];   // A tile: 64 rows x 768 bf16 (padded)  ~97 KB
    float          h[64 * HP];   // h tile for epilogue                  ~98 KB
};

__global__ __launch_bounds__(256)
void gemm_ln_gelu_kernel(const unsigned short* __restrict__ pooledT,
                         const unsigned short* __restrict__ w1t,
                         const float* __restrict__ b1,
                         const float* __restrict__ gamma,
                         const float* __restrict__ beta,
                         const float* __restrict__ W2,
                         const float* __restrict__ b2,
                         const float* __restrict__ counts,
                         float* __restrict__ out) {
    const int d    = blockIdx.y;
    const int b0   = blockIdx.x * 64;       // 64 batch rows per block
    const int tid  = threadIdx.x;
    const int wave = tid >> 5;
    const int lane = tid & 31;
    const int l16  = lane & 15;
    const int kb   = (lane < 16) ? 0 : 8;   // ISA 16-bit fragment K split

    __shared__ SmemU sm;

    // ---- Stage A tile (pooledT[d][b0..b0+63][:]) into LDS, once ----
    {
        const unsigned short* Ag = pooledT + ((size_t)d * Bdim + b0) * Hdim;
        for (int idx = tid; idx < 64 * (Hdim / 8); idx += 256) {   // 64 x 96 uint4
            const int r = idx / (Hdim / 8), c = idx % (Hdim / 8);
            uint4 v = *(const uint4*)(Ag + (size_t)r * Hdim + c * 8);
            *(uint4*)(&sm.a[r * AP + c * 8]) = v;
        }
    }
    __syncthreads();

    // B rows: w1t[d][n][:] where n = 48*wave + 16*t + l16
    const unsigned short* Bp0 = w1t + ((size_t)d * Kdim + (48 * wave + l16)) * Hdim;
    const unsigned short* Bp1 = Bp0 + (size_t)16 * Hdim;
    const unsigned short* Bp2 = Bp0 + (size_t)32 * Hdim;

    v8f acc[4][3];
#pragma unroll
    for (int m = 0; m < 4; ++m)
#pragma unroll
        for (int t = 0; t < 3; ++t)
#pragma unroll
            for (int v = 0; v < 8; ++v) acc[m][t][v] = 0.0f;

    union Frag { v16bf v; uint4 q[2]; };

    for (int k = 0; k < Hdim; k += 32) {
        Frag bm[3];
        bm[0].q[0] = *(const uint4*)(Bp0 + k + kb);
        bm[0].q[1] = *(const uint4*)(Bp0 + k + kb + 16);
        bm[1].q[0] = *(const uint4*)(Bp1 + k + kb);
        bm[1].q[1] = *(const uint4*)(Bp1 + k + kb + 16);
        bm[2].q[0] = *(const uint4*)(Bp2 + k + kb);
        bm[2].q[1] = *(const uint4*)(Bp2 + k + kb + 16);
        Frag a[4];
#pragma unroll
        for (int m = 0; m < 4; ++m) {
            const unsigned short* as = &sm.a[(16 * m + l16) * AP + k + kb];
            a[m].q[0] = *(const uint4*)(as);
            a[m].q[1] = *(const uint4*)(as + 16);
        }
#pragma unroll
        for (int m = 0; m < 4; ++m)
#pragma unroll
            for (int t = 0; t < 3; ++t)
                acc[m][t] = __builtin_amdgcn_wmma_f32_16x16x32_bf16(
                    false, a[m].v, false, bm[t].v,
                    (short)0, acc[m][t], false, false);
    }

    __syncthreads();   // all waves done reading sm.a before we overwrite as sm.h

    // Spill tile: VGPR v -> M = 16*m + v + (lane>=16 ? 8 : 0), N = 48w+16t+l16
    const int mhi = (lane >= 16) ? 8 : 0;
#pragma unroll
    for (int m = 0; m < 4; ++m)
#pragma unroll
        for (int t = 0; t < 3; ++t) {
            const int ncol = 48 * wave + 16 * t + l16;
#pragma unroll
            for (int v = 0; v < 8; ++v)
                sm.h[(16 * m + mhi + v) * HP + ncol] = acc[m][t][v];
        }
    __syncthreads();

    // Epilogue: 4 threads per row (64 rows x 4 = 256); each owns 96 columns.
    const int row  = tid >> 2;
    const int sub  = tid & 3;
    const int base = sub * 96;
    const float* g_  = gamma + (size_t)d * Kdim;
    const float* be_ = beta  + (size_t)d * Kdim;
    const float* w2_ = W2    + (size_t)d * Kdim;
    const float* b1_ = b1    + (size_t)d * Kdim;
    const float* hrow = &sm.h[row * HP];

    float s = 0.0f, s2 = 0.0f;
#pragma unroll 8
    for (int i = 0; i < 96; ++i) {
        float v = hrow[base + i] + b1_[base + i];
        s += v; s2 += v * v;
    }
#pragma unroll
    for (int off = 1; off < 4; off <<= 1) {
        s  += __shfl_xor(s,  off, 4);
        s2 += __shfl_xor(s2, off, 4);
    }
    const float mu   = s * (1.0f / Kdim);
    const float var  = s2 * (1.0f / Kdim) - mu * mu;
    const float rsig = rsqrtf(var + LN_EPS);

    float dot = 0.0f;
#pragma unroll 8
    for (int i = 0; i < 96; ++i) {
        float v  = hrow[base + i] + b1_[base + i];
        float hn = (v - mu) * rsig * g_[base + i] + be_[base + i];
        float ge = 0.5f * hn * (1.0f + erff(hn * 0.70710678118654752f));
        dot += ge * w2_[base + i];
    }
#pragma unroll
    for (int off = 1; off < 4; off <<= 1) dot += __shfl_xor(dot, off, 4);

    if (sub == 0) {
        const int b = b0 + row;
        float r = (counts[d] > 0.0f) ? (dot + b2[d]) : 0.0f;
        out[(size_t)b * Ddim + d] = r;
    }
}

// ---------------- Launch ----------------
extern "C" void kernel_launch(void* const* d_in, const int* in_sizes, int n_in,
                              void* d_out, int out_size, void* d_ws, size_t ws_size,
                              hipStream_t stream) {
    (void)in_sizes; (void)n_in; (void)out_size; (void)ws_size;
    const float* rf    = (const float*)d_in[0];
    const int*   mask  = (const int*)d_in[1];
    const float* W1    = (const float*)d_in[2];
    const float* b1    = (const float*)d_in[3];
    const float* gamma = (const float*)d_in[4];
    const float* beta  = (const float*)d_in[5];
    const float* W2    = (const float*)d_in[6];
    const float* b2    = (const float*)d_in[7];
    float* out = (float*)d_out;

    char* ws = (char*)d_ws;
    const size_t POOLED_BYTES = (size_t)Ddim * Bdim * Hdim * 2;  // 88,080,384
    const size_t W1T_BYTES    = (size_t)Ddim * Kdim * Hdim * 2;  //  8,257,536
    unsigned short* pooledT = (unsigned short*)ws;
    unsigned short* w1t     = (unsigned short*)(ws + POOLED_BYTES);
    float* counts = (float*)(ws + POOLED_BYTES + W1T_BYTES);
    float* rdenom = counts + 16;

    counts_kernel<<<1, 32, 0, stream>>>(mask, counts, rdenom);
    w1t_kernel<<<Ddim * Kdim, 256, 0, stream>>>(W1, w1t);
    pool_kernel<<<dim3(Hdim / 256, Bdim), 256, 0, stream>>>(rf, mask, rdenom, pooledT);
    gemm_ln_gelu_kernel<<<dim3(Bdim / 64, Ddim), 256, 0, stream>>>(
        pooledT, w1t, b1, gamma, beta, W2, b2, counts, out);
}
